// AFNOSpectralAttention_12429635354966
// MI455X (gfx1250) — matrix operand order repaired and checked by reading, not compiled, not run on hardware
//
#include <hip/hip_runtime.h>

typedef unsigned short u16;
typedef __attribute__((ext_vector_type(16))) __bf16 v16bf;
typedef __attribute__((ext_vector_type(8)))  float  v8f;

namespace {
constexpr int kH  = 224;
constexpr int kW  = 224;
constexpr int kC  = 768;
constexpr int kNB = 8;
constexpr int kBS = 96;
constexpr int kM  = 96;     // modified W-frequencies
constexpr int kF  = 113;    // rfft length = W/2+1
constexpr int kB  = 2;
constexpr int kHW = kH * kW;          // 50176 tokens / image
constexpr int kTT = kW * kW;          // one T matrix: 224*224
constexpr int kLdsStride = 232;       // 224+8 halves -> 464B row stride (16B multiple)
constexpr float kTw     = 6.28318530717958647692f / 224.0f;  // 2*pi/W
constexpr float kInvSqW = 0.06681531047810609f;              // 1/sqrt(224)
}

// ---------- small helpers ----------
struct alignas(16) Q128 { unsigned x, y, z, w; };
union F16x16 { Q128 q[2]; v16bf v; };

// A fragment (16x32 bf16): lane half h, row = tile_m*16 + (lane&15).
// lane<16 holds K{0..7,16..23}; lane>=16 holds K{8..15,24..31}.
__device__ __forceinline__ v16bf ldfragA(const u16* p) {   // p = row_base + ktile*32 + 8*half
  F16x16 f;
  f.q[0] = *(const Q128*)(p);
  f.q[1] = *(const Q128*)(p + 16);
  return f.v;
}
// B fragment (32x16 bf16): lane half h, col = tile_n*16 + (lane&15); lane<16 holds K{0..7,16..23}; lane>=16 holds K{8..15,24..31}.
__device__ __forceinline__ v16bf ldfragB(const u16* p) {   // p = col_base + ktile*32 + 8*half
  F16x16 f;
  f.q[0] = *(const Q128*)(p);
  f.q[1] = *(const Q128*)(p + 16);
  return f.v;
}

__device__ __forceinline__ v8f wmma_bf16(v16bf a, v16bf b, v8f c) {
  return __builtin_amdgcn_wmma_f32_16x16x32_bf16(false, a, false, b, (short)0, c,
                                                 false, false);
}

__device__ __forceinline__ u16 f2bf(float f) {             // RNE f32 -> bf16
  unsigned u = __float_as_uint(f);
  unsigned r = (u + 0x7fffu + ((u >> 16) & 1u)) >> 16;
  return (u16)r;
}
__device__ __forceinline__ float bf2f(u16 h) { return __uint_as_float(((unsigned)h) << 16); }

__device__ __forceinline__ float sigmoidf_(float x) { return 1.0f / (1.0f + __expf(-x)); }

// ---------- precompute kernels ----------
// G[blk][k][t] (complex): column t of (L_b * rfft_matrix), k = 0..112
__global__ void k_build_G(const float* __restrict__ bw, const float* __restrict__ gates,
                          float2* __restrict__ G) {
  int id = blockIdx.x * 256 + threadIdx.x;
  if (id >= kNB * kF * kW) return;
  int blk = id / (kF * kW);
  int rem = id % (kF * kW);
  int k = rem / kW, t = rem % kW;
  float re = 0.f, im = 0.f;
  if (k < kM) {
    float g = sigmoidf_(gates[blk]);
    const float* wcol = bw + (size_t)blk * kBS * kBS + k;   // W[blk][m][k], stride 96 over m
    for (int m = 0; m < kM; ++m) {
      float w = wcol[(size_t)m * kBS];
      float ang = -(float)((m * t) % kW) * kTw;
      float s, c;
      sincosf(ang, &s, &c);
      re += w * c;
      im += w * s;
    }
    re *= g * kInvSqW;
    im *= g * kInvSqW;
  } else {
    float ang = -(float)((k * t) % kW) * kTw;
    float s, c;
    sincosf(ang, &s, &c);
    re = c * kInvSqW;
    im = s * kInvSqW;
  }
  G[id] = make_float2(re, im);
}

// Tt[blk][t'][t] = T_b[t][t'] (A-matrix layout for out = T^T @ X), split to bf16 hi/lo
__global__ void k_build_T(const float2* __restrict__ G, u16* __restrict__ Th,
                          u16* __restrict__ Tl) {
  int id = blockIdx.x * 256 + threadIdx.x;
  if (id >= kNB * kTT) return;
  int blk = id / kTT;
  int rem = id % kTT;
  int t = rem / kW, tp = rem % kW;
  const float2* Gp = G + (size_t)blk * kF * kW + t;
  float acc = 0.f;
  for (int k = 0; k < kF; ++k) {
    float2 gk = Gp[(size_t)k * kW];
    float ang = (float)((k * tp) % kW) * kTw;
    float s, c;
    sincosf(ang, &s, &c);
    float beta = (k == 0 || k == kF - 1) ? 1.f : 2.f;
    acc += beta * (gk.x * c - gk.y * s);
  }
  acc *= kInvSqW;
  size_t o = (size_t)blk * kTT + (size_t)tp * kW + t;
  u16 hi = f2bf(acc);
  Th[o] = hi;
  Tl[o] = f2bf(acc - bf2f(hi));
}

// brow[blk][t'] = sqrt(H) * irfft_W(g*bias)[t']  (added only at spatial y==0)
__global__ void k_build_brow(const float* __restrict__ bias, const float* __restrict__ gates,
                             float* __restrict__ brow) {
  int id = blockIdx.x * 256 + threadIdx.x;
  if (id >= kNB * kW) return;
  int blk = id / kW, tp = id % kW;
  float g = sigmoidf_(gates[blk]);
  const float* bb = bias + (size_t)blk * kBS;
  float acc = bb[0];
  for (int k = 1; k < kM; ++k)
    acc += 2.f * bb[k] * __cosf((float)((k * tp) % kW) * kTw);
  brow[id] = g * acc;   // sqrt(H)/sqrt(W) == 1 for H==W
}

// Mpt[c][r] = rescale*proj_w[c][r] + (r==c)   (projection + residual folded), bf16 hi/lo
__global__ void k_build_Mpt(const float* __restrict__ pw, const float* __restrict__ rescale,
                            u16* __restrict__ Mh, u16* __restrict__ Ml) {
  int id = blockIdx.x * 256 + threadIdx.x;
  if (id >= kC * kC) return;
  int c = id / kC, r = id % kC;
  float v = rescale[0] * pw[id] + ((r == c) ? 1.f : 0.f);
  u16 hi = f2bf(v);
  Mh[id] = hi;
  Ml[id] = f2bf(v - bf2f(hi));
}

// ---------- pass 2: spectral rows  out[224 x 96] = T_b^T @ X  per (b,y,blk) ----------
__global__ __launch_bounds__(256) void k_spectral(const float* __restrict__ x,
                                                  const u16* __restrict__ Th,
                                                  const u16* __restrict__ Tl,
                                                  const float* __restrict__ brow,
                                                  u16* __restrict__ xoh,
                                                  u16* __restrict__ xol) {
  extern __shared__ u16 smem[];
  u16* Xh = smem;                       // [96][232] bf16-hi, layout [c][t]
  u16* Xl = smem + kBS * kLdsStride;    // [96][232] bf16-lo
  const int y = blockIdx.x, blk = blockIdx.y, b = blockIdx.z;

  // stage 224(t) x 96(c) fp32 -> LDS bf16 hi/lo, transposed to [c][t]
  const float* xs = x + (((size_t)b * kHW) + (size_t)y * kW) * kC + (size_t)blk * kBS;
  for (int i = threadIdx.x; i < kW * kBS; i += 256) {
    int t = i / kBS, c = i % kBS;
    float f = xs[(size_t)t * kC + c];
    u16 hi = f2bf(f);
    Xh[c * kLdsStride + t] = hi;
    Xl[c * kLdsStride + t] = f2bf(f - bf2f(hi));
  }
  __syncthreads();

  const int wave = threadIdx.x >> 5, lane = threadIdx.x & 31;
  const int half = lane >> 4, idx = lane & 15;
  const u16* Tbh = Th + (size_t)blk * kTT;
  const u16* Tbl = Tl + (size_t)blk * kTT;

  for (int tile = wave; tile < 14 * 6; tile += 8) {   // 14 m-tiles x 6 n-tiles
    const int mt = tile % 14, nt = tile / 14;
    v8f acc;
    #pragma unroll
    for (int i = 0; i < 8; ++i) acc[i] = 0.f;

    const u16* Arh = Tbh + (size_t)(mt * 16 + idx) * kW;
    const u16* Arl = Tbl + (size_t)(mt * 16 + idx) * kW;
    const u16* Bch = Xh + (nt * 16 + idx) * kLdsStride;
    const u16* Bcl = Xl + (nt * 16 + idx) * kLdsStride;

    #pragma unroll
    for (int kt = 0; kt < 7; ++kt) {                  // K = 224 = 7 x 32
      const int ka = kt * 32 + 8 * half;
      const int kb = kt * 32 + 8 * half;
      v16bf ah = ldfragA(Arh + ka);
      v16bf al = ldfragA(Arl + ka);
      v16bf bh = ldfragB(Bch + kb);
      v16bf bl = ldfragB(Bcl + kb);
      acc = wmma_bf16(ah, bh, acc);   // hi*hi
      acc = wmma_bf16(ah, bl, acc);   // hi*lo
      acc = wmma_bf16(al, bh, acc);   // lo*hi  (lo*lo negligible)
    }

    // epilogue: +bias row at y==0, split to bf16 hi/lo intermediate
    const int ch = blk * kBS + nt * 16 + idx;
    #pragma unroll
    for (int v = 0; v < 8; ++v) {
      const int tp = mt * 16 + v + 8 * half;
      float val = acc[v];
      if (y == 0) val += brow[blk * kW + tp];
      size_t o = (((size_t)b * kHW) + (size_t)y * kW + tp) * kC + ch;
      u16 hi = f2bf(val);
      xoh[o] = hi;
      xol[o] = f2bf(val - bf2f(hi));
    }
  }
}

// ---------- pass 3: final = x_out @ (rescale*W^T + I) + rescale*proj_b ----------
__global__ __launch_bounds__(256) void k_proj(const u16* __restrict__ xoh,
                                              const u16* __restrict__ xol,
                                              const u16* __restrict__ Mh,
                                              const u16* __restrict__ Ml,
                                              const float* __restrict__ pb,
                                              const float* __restrict__ rescale,
                                              float* __restrict__ out) {
  const int wave = threadIdx.x >> 5, lane = threadIdx.x & 31;
  const int half = lane >> 4, idx = lane & 15;
  const int mtile = blockIdx.x * 8 + wave;            // 6272 m-tiles total
  const int cbase = blockIdx.y * 96;                  // 8 column groups of 96

  const size_t row = (size_t)mtile * 16 + idx;
  const u16* Arh = xoh + row * kC;
  const u16* Arl = xol + row * kC;

  v8f acc[6];
  #pragma unroll
  for (int n = 0; n < 6; ++n)
    #pragma unroll
    for (int i = 0; i < 8; ++i) acc[n][i] = 0.f;

  for (int kt = 0; kt < kC / 32; ++kt) {              // 24 K-steps
    const int ka = kt * 32 + 8 * half;
    const int kb = kt * 32 + 8 * half;
    if (kt < 23) __builtin_prefetch(Arh + ka + 32, 0, 1);   // global_prefetch_b8
    v16bf ah = ldfragA(Arh + ka);
    v16bf al = ldfragA(Arl + ka);
    #pragma unroll
    for (int nt = 0; nt < 6; ++nt) {
      const size_t bo = (size_t)(cbase + nt * 16 + idx) * kC + kb;
      v16bf bh = ldfragB(Mh + bo);
      v16bf bl = ldfragB(Ml + bo);
      acc[nt] = wmma_bf16(ah, bh, acc[nt]);
      acc[nt] = wmma_bf16(ah, bl, acc[nt]);
      acc[nt] = wmma_bf16(al, bh, acc[nt]);
    }
  }

  const float rs = rescale[0];
  #pragma unroll
  for (int nt = 0; nt < 6; ++nt) {
    const int c = cbase + nt * 16 + idx;
    const float pbias = rs * pb[c];
    #pragma unroll
    for (int v = 0; v < 8; ++v) {
      const size_t tok = (size_t)mtile * 16 + v + 8 * half;
      out[tok * kC + c] = acc[nt][v] + pbias;
    }
  }
}

// ---------- launcher ----------
extern "C" void kernel_launch(void* const* d_in, const int* in_sizes, int n_in,
                              void* d_out, int out_size, void* d_ws, size_t ws_size,
                              hipStream_t stream) {
  const float* x      = (const float*)d_in[0];
  const float* bw     = (const float*)d_in[1];
  const float* bbias  = (const float*)d_in[2];
  const float* gates  = (const float*)d_in[3];
  const float* pw     = (const float*)d_in[4];
  const float* pb     = (const float*)d_in[5];
  const float* resc   = (const float*)d_in[6];
  float* out = (float*)d_out;

  // workspace layout (all offsets 16B aligned)
  char* ws = (char*)d_ws;
  constexpr size_t szT   = (size_t)kNB * kTT * sizeof(u16);        //   802,816
  constexpr size_t szG   = (size_t)kNB * kF * kW * sizeof(float2); // 1,619,968
  constexpr size_t szBr  = (size_t)kNB * kW * sizeof(float);       //     7,168
  constexpr size_t szM   = (size_t)kC * kC * sizeof(u16);          // 1,179,648
  constexpr size_t szXo  = (size_t)kB * kHW * kC * sizeof(u16);    // 154,140,672
  size_t o = 0;
  u16*    Th   = (u16*)   (ws + o); o += szT;
  u16*    Tl   = (u16*)   (ws + o); o += szT;
  float2* G    = (float2*)(ws + o); o += szG;
  float*  brow = (float*) (ws + o); o += szBr;
  u16*    Mh   = (u16*)   (ws + o); o += szM;
  u16*    Ml   = (u16*)   (ws + o); o += szM;
  u16*    Xoh  = (u16*)   (ws + o); o += szXo;
  u16*    Xol  = (u16*)   (ws + o); o += szXo;   // total ~314 MB
  (void)ws_size; (void)in_sizes; (void)n_in; (void)out_size;

  // precompute (tiny, L2-resident results)
  k_build_G   <<<(kNB * kF * kW + 255) / 256, 256, 0, stream>>>(bw, gates, G);
  k_build_T   <<<(kNB * kTT + 255) / 256,     256, 0, stream>>>(G, Th, Tl);
  k_build_brow<<<(kNB * kW + 255) / 256,      256, 0, stream>>>(bbias, gates, brow);
  k_build_Mpt <<<(kC * kC + 255) / 256,       256, 0, stream>>>(pw, resc, Mh, Ml);

  // pass 2: spectral operator along W rows (WMMA, X tile staged in LDS)
  const size_t lds = 2u * kBS * kLdsStride * sizeof(u16);  // 89,088 B
  k_spectral<<<dim3(kH, kNB, kB), 256, lds, stream>>>(x, Th, Tl, brow, Xoh, Xol);

  // pass 3: folded projection + residual GEMM (WMMA)
  k_proj<<<dim3((kB * kHW / 16) / 8, kC / 96), 256, 0, stream>>>(Xoh, Xol, Mh, Ml,
                                                                 pb, resc, out);
}
